// SioConvLayer_19722489823470
// MI455X (gfx1250) — matrix operand order repaired and checked by reading, not compiled, not run on hardware
//
#include <hip/hip_runtime.h>
#include <hip/hip_bf16.h>

// ---------------------------------------------------------------------------
// SioConv layer for MI455X (gfx1250, wave32, WMMA f32_16x16x32_f16)
//   B=4, L=1024, DIM=1024, H=8, I=64 (complex), d2 = H*2I = 1024
// v4: double-buffered async-DMA K-loop (BK=64) in the dense GEMMs so
//     global_load_async_to_lds_b128 overlaps WMMA; all operand fragments are
//     2x ds_load_b128; f16 operand planes precomputed (weights transposed).
// ---------------------------------------------------------------------------

typedef __attribute__((ext_vector_type(16))) _Float16 v16h;
typedef __attribute__((ext_vector_type(8)))  _Float16 v8h;
typedef __attribute__((ext_vector_type(8)))  float    v8f;
typedef int v4i __attribute__((__vector_size__(16)));
typedef __attribute__((address_space(1))) v4i* gp4;   // global int4*
typedef __attribute__((address_space(3))) v4i* lp4;   // LDS int4*

#define LOG_A   (-1.0000050000287824e-05f)   /* logf(1 - 1e-5) */
#define PANG_C  (-0.14391156516f)            /* logf(1e-4)/64  */

#if __has_builtin(__builtin_amdgcn_global_load_async_to_lds_b128)
#define HAVE_ASYNC 1
#else
#define HAVE_ASYNC 0
#endif

// 16-byte global -> LDS copy (async DMA when available)
__device__ __forceinline__ void cp16(_Float16* lds, const _Float16* g) {
#if HAVE_ASYNC
  __builtin_amdgcn_global_load_async_to_lds_b128((gp4)g, (lp4)lds, 0, 0);
#else
  *(v8h*)lds = *(const v8h*)g;
#endif
}

template <int N>
__device__ __forceinline__ void async_wait_le() {
#if HAVE_ASYNC
#if __has_builtin(__builtin_amdgcn_s_wait_asynccnt)
  __builtin_amdgcn_s_wait_asynccnt(N);
#else
  asm volatile("s_wait_asynccnt %0" ::"i"(N) : "memory");
#endif
#endif
}

__device__ __forceinline__ v8f vzero8() {
  v8f v;
#pragma unroll
  for (int i = 0; i < 8; ++i) v[i] = 0.f;
  return v;
}

__device__ __forceinline__ v8f wmma32(v16h a, v16h b, v8f c) {
  return __builtin_amdgcn_wmma_f32_16x16x32_f16(false, a, false, b, (short)0, c,
                                                false, false);
}

// A fragment (16x32 f16), row-major [M][K] LDS tile; 2x ds_load_b128.
__device__ __forceinline__ v16h load_a_frag(const _Float16* T, int stride,
                                            int row0, int k0, int lane) {
  const int row = row0 + (lane & 15);
  const int kb  = k0 + ((lane >> 4) << 3);
  const _Float16* p = T + row * stride + kb;
  v8h lo = *(const v8h*)p;
  v8h hi = *(const v8h*)(p + 16);
  return __builtin_shufflevector(lo, hi, 0, 1, 2, 3, 4, 5, 6, 7,
                                 8, 9, 10, 11, 12, 13, 14, 15);
}

// B fragment (32x16 f16) from [N][K] LDS tile (K contiguous); 2x ds_load_b128.
__device__ __forceinline__ v16h load_b_frag(const _Float16* T, int stride,
                                            int k0, int col0, int lane) {
  const int col = col0 + (lane & 15);
  const int kb  = k0 + ((lane >> 4) << 3);
  const _Float16* p = T + col * stride + kb;
  v8h lo = *(const v8h*)p;
  v8h hi = *(const v8h*)(p + 16);
  return __builtin_shufflevector(lo, hi, 0, 1, 2, 3, 4, 5, 6, 7,
                                 8, 9, 10, 11, 12, 13, 14, 15);
}

// ---------------------------------------------------------------------------
// f32 -> f16 plane conversion (8 elements/thread, b128 in/out)
// ---------------------------------------------------------------------------
__global__ __launch_bounds__(256) void k_cvt(const float* __restrict__ s,
                                             _Float16* __restrict__ d, int n8) {
  int i = blockIdx.x * 256 + threadIdx.x;
  if (i >= n8) return;
  const float4* s4 = (const float4*)s;
  float4 a = s4[i * 2], b = s4[i * 2 + 1];
  v8h o;
  o[0] = (_Float16)a.x; o[1] = (_Float16)a.y; o[2] = (_Float16)a.z; o[3] = (_Float16)a.w;
  o[4] = (_Float16)b.x; o[5] = (_Float16)b.y; o[6] = (_Float16)b.z; o[7] = (_Float16)b.w;
  *(v8h*)(d + (size_t)i * 8) = o;
}

// f32 (K,N)=(1024,1024) weight -> f16 transposed plane whT[n][k]
__global__ __launch_bounds__(256) void k_cvtT(const float* __restrict__ s,
                                              _Float16* __restrict__ d) {
  int idx = blockIdx.x * 256 + threadIdx.x;   // 131072 = 1024 * 128
  int n = idx >> 7, k8 = (idx & 127) << 3;
  v8h o;
#pragma unroll
  for (int e = 0; e < 8; ++e) o[e] = (_Float16)s[(size_t)(k8 + e) * 1024 + n];
  *(v8h*)(d + (size_t)n * 1024 + k8) = o;
}

// ---------------------------------------------------------------------------
// a_angle = x @ Wa + ba  (kept entirely fp32: its error is amplified by (l-m))
// ---------------------------------------------------------------------------
__global__ __launch_bounds__(256) void k_aproj(const float* __restrict__ x,
                                               const float* __restrict__ Wa,
                                               const float* __restrict__ ba,
                                               float* __restrict__ a_ws) {
  const int idx = blockIdx.x * 256 + threadIdx.x;   // 32768 = 4096*8
  const int row = idx >> 3, col = idx & 7;
  float acc = 0.f;
  for (int k = 0; k < 1024; ++k) {
    if ((k & 63) == 0) __builtin_prefetch(&x[(size_t)row * 1024 + k + 64], 0, 0);
    acc = fmaf(x[(size_t)row * 1024 + k], Wa[(size_t)k * 8 + col], acc);
  }
  a_ws[idx] = acc + ba[col];
}

// ---------------------------------------------------------------------------
// Fused q/k/v/g projection GEMM; double-buffered async tiles, BK=64.
// Outputs: q_re/q_im (B,H,L,I), k_re/k_im (B,H,L,I), kT_re/kT_im (B,H,I,L),
//          vT_re/vT_im (B,H,I,L), g (B*L,1024) f32.
// ---------------------------------------------------------------------------
__global__ __launch_bounds__(256) void k_proj(
    const _Float16* __restrict__ xh, const _Float16* __restrict__ whT,
    const float* __restrict__ bq, const float* __restrict__ bk,
    const float* __restrict__ bv, const float* __restrict__ bg,
    _Float16* __restrict__ qre, _Float16* __restrict__ qim,
    _Float16* __restrict__ kre, _Float16* __restrict__ kim,
    _Float16* __restrict__ kTre, _Float16* __restrict__ kTim,
    _Float16* __restrict__ vTre, _Float16* __restrict__ vTim,
    float* __restrict__ g_ws) {
  __shared__ __attribute__((aligned(16))) _Float16 XA[2][64][72];  // [M][K]
  __shared__ __attribute__((aligned(16))) _Float16 WB[2][64][72];  // [N][K]
  const int tid = threadIdx.x, lane = tid & 31, wave = tid >> 5;
  const int nbase = blockIdx.x * 64;
  const int mbase = blockIdx.y * 64;
  const int which = nbase >> 10;          // 0:q 1:k 2:v 3:g
  const int nloc  = nbase & 1023;
  const _Float16* W = whT + (size_t)which * 1024 * 1024;
  const float* bias = (which == 0) ? bq : (which == 1) ? bk : (which == 2) ? bv : bg;

  const int wrow = (wave >> 1) << 4;
  const int wcol = (wave & 1) << 5;
  v8f acc[2]; acc[0] = vzero8(); acc[1] = vzero8();

  const int sr = tid >> 2, sc8 = (tid & 3) << 3;   // chunk coords (64x64: 2 per thr)
  auto stage = [&](int buf, int k0) {
#pragma unroll
    for (int j = 0; j < 2; ++j) {
      int r = sr + 32 * j;   // wait: 512 chunks -> r = (tid+256j)>>3
      (void)r;
    }
  };
  (void)stage;

#define PROJ_STAGE(buf, k0)                                                  \
  {                                                                          \
    _Pragma("unroll") for (int j = 0; j < 2; ++j) {                          \
      int c = tid + 256 * j;                                                 \
      int r = c >> 3, c8 = (c & 7) << 3;                                     \
      cp16(&XA[buf][r][c8], xh + (size_t)(mbase + r) * 1024 + (k0) + c8);    \
      cp16(&WB[buf][r][c8], W + (size_t)(nloc + r) * 1024 + (k0) + c8);      \
    }                                                                        \
  }

  PROJ_STAGE(0, 0);
  for (int k0 = 0; k0 < 1024; k0 += 64) {
    const int cur = (k0 >> 6) & 1;
    if (k0 + 64 < 1024) {
      PROJ_STAGE(cur ^ 1, k0 + 64);
      async_wait_le<4>();                 // drain previous tile, new 4 in flight
    } else {
      async_wait_le<0>();
    }
    __syncthreads();
#pragma unroll
    for (int kk = 0; kk < 64; kk += 32) {
      v16h a  = load_a_frag(&XA[cur][0][0], 72, wrow, kk, lane);
      v16h b0 = load_b_frag(&WB[cur][0][0], 72, kk, wcol, lane);
      v16h b1 = load_b_frag(&WB[cur][0][0], 72, kk, wcol + 16, lane);
      acc[0] = wmma32(a, b0, acc[0]);
      acc[1] = wmma32(a, b1, acc[1]);
    }
    __syncthreads();
  }

#pragma unroll
  for (int t2 = 0; t2 < 2; ++t2) {
#pragma unroll
    for (int r = 0; r < 8; ++r) {
      const int row = wrow + ((lane >> 4) << 3) + r;
      const int col = wcol + (t2 << 4) + (lane & 15);
      const int gm  = mbase + row;        // b*1024 + l
      const int gn  = nloc + col;
      float val = acc[t2][r] + bias[gn];
      const int h = gn >> 7, isim = gn & 1, i = (gn & 127) >> 1;
      const int bb = gm >> 10, l = gm & 1023;
      const int bh = bb * 8 + h;
      if (which <= 1) {
        float other = __shfl_xor(val, 1, 32);
        float th  = (float)l * __expf(PANG_C * (float)i);
        float cth = cosf(th), sth = sinf(th);
        if (which == 1) sth = -sth;       // k uses conj(p_pow)
        float re = isim ? other : val;
        float im = isim ? val : other;
        float res = isim ? (re * sth + im * cth) : (re * cth - im * sth);
        if (which == 0) {
          size_t p = ((size_t)bh * 1024 + l) * 64 + i;
          (isim ? qim : qre)[p] = (_Float16)res;
        } else {
          size_t p  = ((size_t)bh * 1024 + l) * 64 + i;            // (L,I)
          size_t pT = ((size_t)bh * 64 + i) * 1024 + l;            // (I,L)
          (isim ? kim : kre)[p] = (_Float16)res;
          (isim ? kTim : kTre)[pT] = (_Float16)res;
        }
      } else if (which == 2) {
        size_t pT = ((size_t)bh * 64 + i) * 1024 + l;              // vT (I,L)
        (isim ? vTim : vTre)[pT] = (_Float16)val;
      } else {
        g_ws[(size_t)gm * 1024 + gn] = val;
      }
    }
  }
}

// ---------------------------------------------------------------------------
// d[b,h,l] = exp(cumsum_l ln_a), S[b,h] = exp(sum ln_a). One block per (b,h).
// ---------------------------------------------------------------------------
__global__ __launch_bounds__(256) void k_scan(const float* __restrict__ a_ws,
                                              float* __restrict__ d_ws,
                                              float* __restrict__ s_ws) {
  __shared__ float buf[256];
  const int bh = blockIdx.x, b = bh >> 3, h = bh & 7;
  const int t = threadIdx.x;
  float v[4], s = 0.f;
#pragma unroll
  for (int j = 0; j < 4; ++j) {
    int l = t * 4 + j;
    s += a_ws[(size_t)(b * 1024 + l) * 8 + h];
    v[j] = s;
  }
  buf[t] = s;
  __syncthreads();
  for (int off = 1; off < 256; off <<= 1) {
    float xv = (t >= off) ? buf[t - off] : 0.f;
    __syncthreads();
    buf[t] += xv;
    __syncthreads();
  }
  const float excl = buf[t] - s;
#pragma unroll
  for (int j = 0; j < 4; ++j) {
    int l = t * 4 + j;
    float cim = excl + v[j];
    float mag = __expf((float)(l + 1) * LOG_A);
    float dr = mag * cosf(cim), di = mag * sinf(cim);
    d_ws[((size_t)bh * 1024 + l) * 2 + 0] = dr;
    d_ws[((size_t)bh * 1024 + l) * 2 + 1] = di;
    if (l == 1023) { s_ws[bh * 2] = dr; s_ws[bh * 2 + 1] = di; }
  }
}

// ---------------------------------------------------------------------------
// Flash-style complex decay attention + h_cross + GroupNorm + SiLU gate.
// grid = (16 l-tiles, 32 bh). All LDS tiles [64][72] (b128-aligned rows).
// ---------------------------------------------------------------------------
__global__ __launch_bounds__(256) void k_attn(
    const _Float16* __restrict__ qre, const _Float16* __restrict__ qim,
    const _Float16* __restrict__ kre, const _Float16* __restrict__ kim,
    const _Float16* __restrict__ vTre, const _Float16* __restrict__ vTim,
    const float* __restrict__ g_ws, const float* __restrict__ a_ws,
    const float* __restrict__ d_ws,
    const float* __restrict__ hre_g, const float* __restrict__ him_g,
    const float* __restrict__ gamma, const float* __restrict__ beta,
    _Float16* __restrict__ hnh) {
  __shared__ __attribute__((aligned(16))) _Float16 QR[64][72], QI[64][72]; // A: q[l][i]
  __shared__ __attribute__((aligned(16))) _Float16 KR[64][72], KI[64][72]; // B1: k[m][i]; SC/E A
  __shared__ __attribute__((aligned(16))) _Float16 VR[64][72], VI[64][72]; // B2: vT[i][m]; Hp B
  __shared__ float aang[64];
  __shared__ float red_s[64], red_q[64];

  const int tid = threadIdx.x, lane = tid & 31, wave = tid >> 5;
  const int bh = blockIdx.y, b = bh >> 3, h = bh & 7;
  const int l0 = blockIdx.x * 64;
  const _Float16* qre_p = qre + (size_t)bh * 1024 * 64;
  const _Float16* qim_p = qim + (size_t)bh * 1024 * 64;
  const _Float16* kre_p = kre + (size_t)bh * 1024 * 64;
  const _Float16* kim_p = kim + (size_t)bh * 1024 * 64;
  const _Float16* vre_p = vTre + (size_t)bh * 64 * 1024;
  const _Float16* vim_p = vTim + (size_t)bh * 64 * 1024;

#pragma unroll
  for (int j = 0; j < 2; ++j) {                  // stage q tile (pure copy)
    int c = tid + 256 * j;
    int r = c >> 3, c8 = (c & 7) << 3;
    cp16(&QR[r][c8], qre_p + (size_t)(l0 + r) * 64 + c8);
    cp16(&QI[r][c8], qim_p + (size_t)(l0 + r) * 64 + c8);
  }
  async_wait_le<0>();

  const int wrow  = (wave >> 1) << 4;
  const int wcolc = (wave & 1) << 5;
  v8f hr[2], hi[2];
  hr[0] = vzero8(); hr[1] = vzero8(); hi[0] = vzero8(); hi[1] = vzero8();

  for (int mt = 0; mt <= (int)blockIdx.x; ++mt) {
    const int m0 = mt * 64;
    __syncthreads();                             // LDS reuse guard
#pragma unroll
    for (int j = 0; j < 2; ++j) {
      int c = tid + 256 * j;
      int r = c >> 3, c8 = (c & 7) << 3;
      cp16(&KR[r][c8], kre_p + (size_t)(m0 + r) * 64 + c8);  // k[m][i]
      cp16(&KI[r][c8], kim_p + (size_t)(m0 + r) * 64 + c8);
      cp16(&VR[r][c8], vre_p + (size_t)r * 1024 + m0 + c8);  // vT[i][m]
      cp16(&VI[r][c8], vim_p + (size_t)r * 1024 + m0 + c8);
    }
    if (tid < 64) aang[tid] = a_ws[(size_t)(b * 1024 + m0 + tid) * 8 + h];
    async_wait_le<0>();
    __syncthreads();

    // GEMM1: S = q x k^T (complex); B operand = KR/KI [N=m][K=i]
    v8f sre[2], sim[2], stm[2];
    sre[0] = vzero8(); sre[1] = vzero8();
    sim[0] = vzero8(); sim[1] = vzero8();
    stm[0] = vzero8(); stm[1] = vzero8();
#pragma unroll
    for (int kk = 0; kk < 64; kk += 32) {
      v16h ar = load_a_frag(&QR[0][0], 72, wrow, kk, lane);
      v16h ai = load_a_frag(&QI[0][0], 72, wrow, kk, lane);
#pragma unroll
      for (int t2 = 0; t2 < 2; ++t2) {
        v16h br = load_b_frag(&KR[0][0], 72, kk, wcolc + (t2 << 4), lane);
        v16h bi = load_b_frag(&KI[0][0], 72, kk, wcolc + (t2 << 4), lane);
        sre[t2] = wmma32(ar, br, sre[t2]);
        stm[t2] = wmma32(ai, bi, stm[t2]);
        sim[t2] = wmma32(ar, bi, sim[t2]);
        sim[t2] = wmma32(ai, br, sim[t2]);
      }
    }
    sre[0] -= stm[0]; sre[1] -= stm[1];
    __syncthreads();                             // before overwriting KR/KI

    // decay c[l,m] = exp(ln_a[m]*(l-m)) with causal mask; SC -> KR/KI [l][m]
#pragma unroll
    for (int t2 = 0; t2 < 2; ++t2) {
#pragma unroll
      for (int r = 0; r < 8; ++r) {
        int lrow = wrow + ((lane >> 4) << 3) + r;
        int mcol = wcolc + (t2 << 4) + (lane & 15);
        float dm = (float)((l0 + lrow) - (m0 + mcol));
        float scr = 0.f, sci = 0.f;
        if (dm >= 0.f) {
          float ang = aang[mcol] * dm;
          float mag = __expf(LOG_A * dm);
          float cr = mag * cosf(ang), ci = mag * sinf(ang);
          float a0 = sre[t2][r], b0 = sim[t2][r];
          scr = a0 * cr - b0 * ci;
          sci = a0 * ci + b0 * cr;
        }
        KR[lrow][mcol] = (_Float16)scr;
        KI[lrow][mcol] = (_Float16)sci;
      }
    }
    __syncthreads();

    // GEMM2: h += SC x v; A = KR/KI [l][m], B = VR/VI [N=i][K=m]
    v8f t0 = vzero8(), t1 = vzero8();
#pragma unroll
    for (int kk = 0; kk < 64; kk += 32) {
      v16h ar = load_a_frag(&KR[0][0], 72, wrow, kk, lane);
      v16h ai = load_a_frag(&KI[0][0], 72, wrow, kk, lane);
#pragma unroll
      for (int t2 = 0; t2 < 2; ++t2) {
        v16h br = load_b_frag(&VR[0][0], 72, kk, wcolc + (t2 << 4), lane);
        v16h bi = load_b_frag(&VI[0][0], 72, kk, wcolc + (t2 << 4), lane);
        hr[t2] = wmma32(ar, br, hr[t2]);
        if (t2 == 0) t0 = wmma32(ai, bi, t0); else t1 = wmma32(ai, bi, t1);
        hi[t2] = wmma32(ar, bi, hi[t2]);
        hi[t2] = wmma32(ai, br, hi[t2]);
      }
    }
    hr[0] -= t0; hr[1] -= t1;
  }

  // h_cross: E = q*d -> KR/KI [l][i]; Hp -> VR/VI [N=j][K=i]
  __syncthreads();
#pragma unroll
  for (int j = 0; j < 16; ++j) {
    int idx = tid + 256 * j;
    int r = idx >> 6, c = idx & 63;
    float qr = (float)QR[r][c], qi = (float)QI[r][c];
    float dr = d_ws[((size_t)bh * 1024 + l0 + r) * 2 + 0];
    float di = d_ws[((size_t)bh * 1024 + l0 + r) * 2 + 1];
    KR[r][c] = (_Float16)(qr * dr - qi * di);
    KI[r][c] = (_Float16)(qr * di + qi * dr);
    // element (i=r, j=c) of Hp, stored transposed at VR[c][r]
    float hrv = hre_g[(size_t)bh * 4096 + idx];
    float hiv = him_g[(size_t)bh * 4096 + idx];
    float th = __expf(PANG_C * (float)r);        // p_angle(i=r)
    float ct = cosf(th), st = sinf(th);
    VR[c][r] = (_Float16)(hrv * ct - hiv * st);
    VI[c][r] = (_Float16)(hrv * st + hiv * ct);
  }
  __syncthreads();
  {
    v8f t0 = vzero8(), t1 = vzero8();
#pragma unroll
    for (int kk = 0; kk < 64; kk += 32) {
      v16h ar = load_a_frag(&KR[0][0], 72, wrow, kk, lane);
      v16h ai = load_a_frag(&KI[0][0], 72, wrow, kk, lane);
#pragma unroll
      for (int t2 = 0; t2 < 2; ++t2) {
        v16h br = load_b_frag(&VR[0][0], 72, kk, wcolc + (t2 << 4), lane);
        v16h bi = load_b_frag(&VI[0][0], 72, kk, wcolc + (t2 << 4), lane);
        hr[t2] = wmma32(ar, br, hr[t2]);
        if (t2 == 0) t0 = wmma32(ai, bi, t0); else t1 = wmma32(ai, bi, t1);
        hi[t2] = wmma32(ar, bi, hi[t2]);
        hi[t2] = wmma32(ai, br, hi[t2]);
      }
    }
    hr[0] -= t0; hr[1] -= t1;
  }

  // GroupNorm over 128 re/im features per row + SiLU(g) gate -> f16 plane
  if (tid < 64) { red_s[tid] = 0.f; red_q[tid] = 0.f; }
  __syncthreads();
#pragma unroll
  for (int t2 = 0; t2 < 2; ++t2)
#pragma unroll
    for (int r = 0; r < 8; ++r) {
      int lrow = wrow + ((lane >> 4) << 3) + r;
      float re = hr[t2][r], im = hi[t2][r];
      atomicAdd(&red_s[lrow], re + im);
      atomicAdd(&red_q[lrow], re * re + im * im);
    }
  __syncthreads();
  const float gmm = gamma[h], bet = beta[h];
#pragma unroll
  for (int t2 = 0; t2 < 2; ++t2)
#pragma unroll
    for (int r = 0; r < 8; ++r) {
      int lrow = wrow + ((lane >> 4) << 3) + r;
      int ic   = wcolc + (t2 << 4) + (lane & 15);
      float mean = red_s[lrow] * (1.f / 128.f);
      float var  = red_q[lrow] * (1.f / 128.f) - mean * mean;
      float rstd = rsqrtf(var + 1e-5f);
      float re = hr[t2][r], im = hi[t2][r];
      size_t rb = ((size_t)(b * 1024 + l0 + lrow)) * 1024 + h * 128 + ic * 2;
      float2 gg = *(const float2*)(g_ws + rb);
      float sre_ = gg.x / (1.f + __expf(-gg.x));
      float sim_ = gg.y / (1.f + __expf(-gg.y));
      union { _Float16 hv[2]; unsigned u; } pk;
      pk.hv[0] = (_Float16)(((re - mean) * rstd * gmm + bet) * sre_);
      pk.hv[1] = (_Float16)(((im - mean) * rstd * gmm + bet) * sim_);
      *(unsigned*)(hnh + rb) = pk.u;
    }
}

// ---------------------------------------------------------------------------
// hidden_next: per (b,h) complex GEMM over the full L with decay weights,
// plus rotated cross term; writes d_out part 2 (interleaved complex).
// ---------------------------------------------------------------------------
__global__ __launch_bounds__(256) void k_hnext(
    const _Float16* __restrict__ kTre, const _Float16* __restrict__ kTim,
    const _Float16* __restrict__ vTre, const _Float16* __restrict__ vTim,
    const float* __restrict__ a_ws, const float* __restrict__ s_ws,
    const float* __restrict__ hre_g, const float* __restrict__ him_g,
    float* __restrict__ out2) {
  __shared__ __attribute__((aligned(16))) _Float16 AR[64][72], AI[64][72]; // A: (kT*w)[i][m]
  __shared__ __attribute__((aligned(16))) _Float16 BR[64][72], BI[64][72]; // B: vT[j][m]
  __shared__ float wbr[64], wbi[64];
  const int tid = threadIdx.x, lane = tid & 31, wave = tid >> 5;
  const int bh = blockIdx.x, b = bh >> 3, h = bh & 7;
  const _Float16* kre_p = kTre + (size_t)bh * 64 * 1024;
  const _Float16* kim_p = kTim + (size_t)bh * 64 * 1024;
  const _Float16* vre_p = vTre + (size_t)bh * 64 * 1024;
  const _Float16* vim_p = vTim + (size_t)bh * 64 * 1024;
  const int wrow = (wave >> 1) << 4, wcolc = (wave & 1) << 5;
  v8f hr[2], hi[2];
  hr[0] = vzero8(); hr[1] = vzero8(); hi[0] = vzero8(); hi[1] = vzero8();

  for (int m0 = 0; m0 < 1024; m0 += 64) {
    __syncthreads();
    if (tid < 64) {                              // w[m] = exp(ln_a[m]*(1023-m))
      int gm = m0 + tid;
      float dm = (float)(1023 - gm);
      float av = a_ws[(size_t)(b * 1024 + gm) * 8 + h];
      float mag = __expf(LOG_A * dm);
      wbr[tid] = mag * cosf(av * dm);
      wbi[tid] = mag * sinf(av * dm);
    }
#pragma unroll
    for (int j = 0; j < 2; ++j) {                // vT tiles: pure copies (async)
      int c = tid + 256 * j;
      int r = c >> 3, c8 = (c & 7) << 3;
      cp16(&BR[r][c8], vre_p + (size_t)r * 1024 + m0 + c8);
      cp16(&BI[r][c8], vim_p + (size_t)r * 1024 + m0 + c8);
    }
    __syncthreads();                             // wbr/wbi visible
#pragma unroll
    for (int j = 0; j < 2; ++j) {                // A = kT * w (overlaps DMA)
      int c = tid + 256 * j;
      int r = c >> 3, c8 = (c & 7) << 3;
      v8h krv = *(const v8h*)(kre_p + (size_t)r * 1024 + m0 + c8);
      v8h kiv = *(const v8h*)(kim_p + (size_t)r * 1024 + m0 + c8);
      v8h aro, aio;
#pragma unroll
      for (int e = 0; e < 8; ++e) {
        float wr = wbr[c8 + e], wi = wbi[c8 + e];
        float kr = (float)krv[e], ki = (float)kiv[e];
        aro[e] = (_Float16)(kr * wr - ki * wi);
        aio[e] = (_Float16)(kr * wi + ki * wr);
      }
      *(v8h*)&AR[r][c8] = aro;
      *(v8h*)&AI[r][c8] = aio;
    }
    async_wait_le<0>();
    __syncthreads();
    v8f t0 = vzero8(), t1 = vzero8();
#pragma unroll
    for (int kk = 0; kk < 64; kk += 32) {
      v16h ar = load_a_frag(&AR[0][0], 72, wrow, kk, lane);
      v16h ai = load_a_frag(&AI[0][0], 72, wrow, kk, lane);
#pragma unroll
      for (int t2 = 0; t2 < 2; ++t2) {
        v16h br = load_b_frag(&BR[0][0], 72, kk, wcolc + (t2 << 4), lane);
        v16h bi = load_b_frag(&BI[0][0], 72, kk, wcolc + (t2 << 4), lane);
        hr[t2] = wmma32(ar, br, hr[t2]);
        if (t2 == 0) t0 = wmma32(ai, bi, t0); else t1 = wmma32(ai, bi, t1);
        hi[t2] = wmma32(ar, bi, hi[t2]);
        hi[t2] = wmma32(ai, br, hi[t2]);
      }
    }
    hr[0] -= t0; hr[1] -= t1;
  }

  const float Sr = s_ws[bh * 2], Si = s_ws[bh * 2 + 1];
#pragma unroll
  for (int t2 = 0; t2 < 2; ++t2)
#pragma unroll
    for (int r = 0; r < 8; ++r) {
      int i  = wrow + ((lane >> 4) << 3) + r;
      int jj = wcolc + (t2 << 4) + (lane & 15);
      float th = __expf(PANG_C * (float)i);      // p_angle(i)
      float c1 = cosf(th * 1023.f), s1 = sinf(th * 1023.f);
      float re = hr[t2][r] * c1 - hi[t2][r] * s1;
      float im = hr[t2][r] * s1 + hi[t2][r] * c1;
      float hrv = hre_g[(size_t)bh * 4096 + i * 64 + jj];
      float hiv = him_g[(size_t)bh * 4096 + i * 64 + jj];
      float tre = Sr * hrv - Si * hiv;
      float tim = Sr * hiv + Si * hrv;
      float c2 = cosf(th * 1024.f), s2 = sinf(th * 1024.f);
      re += tre * c2 - tim * s2;
      im += tre * s2 + tim * c2;
      size_t o = ((size_t)bh * 4096 + (size_t)i * 64 + jj) * 2;
      out2[o] = re; out2[o + 1] = im;
    }
}

// ---------------------------------------------------------------------------
// y = hn @ Wy + by (A = f16 plane, B = pre-transposed WyT[n][k]);
// double-buffered async tiles, BK=64.
// ---------------------------------------------------------------------------
__global__ __launch_bounds__(256) void k_ygemm(const _Float16* __restrict__ Ah,
                                               const _Float16* __restrict__ WyT,
                                               const float* __restrict__ by,
                                               float* __restrict__ y) {
  __shared__ __attribute__((aligned(16))) _Float16 XA[2][64][72];
  __shared__ __attribute__((aligned(16))) _Float16 WB[2][64][72];
  const int tid = threadIdx.x, lane = tid & 31, wave = tid >> 5;
  const int nbase = blockIdx.x * 64;
  const int mbase = blockIdx.y * 64;
  const int wrow = (wave >> 1) << 4;
  const int wcol = (wave & 1) << 5;
  v8f acc[2]; acc[0] = vzero8(); acc[1] = vzero8();

#define YG_STAGE(buf, k0)                                                    \
  {                                                                          \
    _Pragma("unroll") for (int j = 0; j < 2; ++j) {                          \
      int c = tid + 256 * j;                                                 \
      int r = c >> 3, c8 = (c & 7) << 3;                                     \
      cp16(&XA[buf][r][c8], Ah + (size_t)(mbase + r) * 1024 + (k0) + c8);    \
      cp16(&WB[buf][r][c8], WyT + (size_t)(nbase + r) * 1024 + (k0) + c8);   \
    }                                                                        \
  }

  YG_STAGE(0, 0);
  for (int k0 = 0; k0 < 1024; k0 += 64) {
    const int cur = (k0 >> 6) & 1;
    if (k0 + 64 < 1024) {
      YG_STAGE(cur ^ 1, k0 + 64);
      async_wait_le<4>();
    } else {
      async_wait_le<0>();
    }
    __syncthreads();
#pragma unroll
    for (int kk = 0; kk < 64; kk += 32) {
      v16h a  = load_a_frag(&XA[cur][0][0], 72, wrow, kk, lane);
      v16h b0 = load_b_frag(&WB[cur][0][0], 72, kk, wcol, lane);
      v16h b1 = load_b_frag(&WB[cur][0][0], 72, kk, wcol + 16, lane);
      acc[0] = wmma32(a, b0, acc[0]);
      acc[1] = wmma32(a, b1, acc[1]);
    }
    __syncthreads();
  }

#pragma unroll
  for (int t2 = 0; t2 < 2; ++t2)
#pragma unroll
    for (int r = 0; r < 8; ++r) {
      int row = wrow + ((lane >> 4) << 3) + r;
      int col = wcol + (t2 << 4) + (lane & 15);
      int gm = mbase + row, gn = nbase + col;
      y[(size_t)gm * 1024 + gn] = acc[t2][r] + by[gn];
    }
}

// ---------------------------------------------------------------------------
extern "C" void kernel_launch(void* const* d_in, const int* in_sizes, int n_in,
                              void* d_out, int out_size, void* d_ws, size_t ws_size,
                              hipStream_t stream) {
  (void)in_sizes; (void)n_in; (void)out_size; (void)ws_size;
  const float* x   = (const float*)d_in[0];
  const float* hre = (const float*)d_in[1];
  const float* him = (const float*)d_in[2];
  const float* Wq = (const float*)d_in[3];  const float* bq = (const float*)d_in[4];
  const float* Wk = (const float*)d_in[5];  const float* bk = (const float*)d_in[6];
  const float* Wv = (const float*)d_in[7];  const float* bv = (const float*)d_in[8];
  const float* Wa = (const float*)d_in[9];  const float* ba = (const float*)d_in[10];
  const float* Wg = (const float*)d_in[11]; const float* bg = (const float*)d_in[12];
  const float* Wy = (const float*)d_in[13]; const float* by = (const float*)d_in[14];
  const float* gamma = (const float*)d_in[15];
  const float* beta  = (const float*)d_in[16];

  float* y_out  = (float*)d_out;                        // (4,1024,1024) f32
  float* hn_out = y_out + (size_t)4 * 1024 * 1024;      // (4,8,64,64) cplx interleaved

  char* p = (char*)d_ws;
  auto carve = [&](size_t bytes) {
    void* r = (void*)p;
    p += (bytes + 255) & ~(size_t)255;
    return r;
  };
  const size_t PL = (size_t)4 * 8 * 1024 * 64;          // plane elements
  _Float16* xh   = (_Float16*)carve((size_t)4096 * 1024 * 2);
  _Float16* whT  = (_Float16*)carve((size_t)5 * 1024 * 1024 * 2); // q,k,v,g,y (transposed)
  _Float16* qre  = (_Float16*)carve(PL * 2);
  _Float16* qim  = (_Float16*)carve(PL * 2);
  _Float16* kre  = (_Float16*)carve(PL * 2);
  _Float16* kim  = (_Float16*)carve(PL * 2);
  _Float16* kTre = (_Float16*)carve(PL * 2);
  _Float16* kTim = (_Float16*)carve(PL * 2);
  _Float16* vTre = (_Float16*)carve(PL * 2);
  _Float16* vTim = (_Float16*)carve(PL * 2);
  float*    g_ws = (float*)carve((size_t)4096 * 1024 * 4);
  float*    a_ws = (float*)carve((size_t)4096 * 8 * 4);
  float*    d_wsp = (float*)carve((size_t)32 * 1024 * 2 * 4);
  float*    s_ws = (float*)carve(64 * 4);
  _Float16* hnh  = (_Float16*)carve((size_t)4096 * 1024 * 2);

  // one-time f16 conversions: x plain, weights transposed
  k_cvt<<<2048, 256, 0, stream>>>(x, xh, 524288);
  k_cvtT<<<512, 256, 0, stream>>>(Wq, whT + (size_t)0 * 1048576);
  k_cvtT<<<512, 256, 0, stream>>>(Wk, whT + (size_t)1 * 1048576);
  k_cvtT<<<512, 256, 0, stream>>>(Wv, whT + (size_t)2 * 1048576);
  k_cvtT<<<512, 256, 0, stream>>>(Wg, whT + (size_t)3 * 1048576);
  k_cvtT<<<512, 256, 0, stream>>>(Wy, whT + (size_t)4 * 1048576);

  k_aproj<<<128, 256, 0, stream>>>(x, Wa, ba, a_ws);
  k_proj<<<dim3(64, 64), 256, 0, stream>>>(xh, whT, bq, bk, bv, bg,
                                           qre, qim, kre, kim, kTre, kTim,
                                           vTre, vTim, g_ws);
  k_scan<<<32, 256, 0, stream>>>(a_ws, d_wsp, s_ws);
  k_attn<<<dim3(16, 32), 256, 0, stream>>>(qre, qim, kre, kim, vTre, vTim,
                                           g_ws, a_ws, d_wsp, hre, him,
                                           gamma, beta, hnh);
  k_hnext<<<32, 256, 0, stream>>>(kTre, kTim, vTre, vTim, a_ws, s_ws,
                                  hre, him, hn_out);
  k_ygemm<<<dim3(16, 64), 256, 0, stream>>>(hnh, whT + (size_t)4 * 1048576, by, y_out);
}
